// BlockModel_28071906246883
// MI455X (gfx1250) — compile-verified, matrix operand order checked by reading, not run on hardware
//
#include <hip/hip_runtime.h>

// ---------------------------------------------------------------------------
// Fused graph-block model for MI455X (gfx1250, wave32, WMMA 16x16x32 bf16)
//   x1 = interp + relu(concat(interp[nidx], add)@Wb0+bb0) segmax @ Wo0 + bo0
//   x2 = x1     + relu(concat(x1[nidx],   add)@Wb1+bb1) segmax @ Wo1 + bo1
//   out = x2 @ Wf + bf
// ---------------------------------------------------------------------------

#define D 128
#define A 4
#define H 128
#define KST_E 168         // LDS k-stride (edge weights): 20-bank step, conflict free
#define KST_P 136         // LDS k-stride (proj weights): 4-bank step, conflict free

typedef __attribute__((ext_vector_type(16))) __bf16 bf16x16;
typedef __attribute__((ext_vector_type(8)))  float  f32x8;

struct alignas(16) U4 { unsigned int x, y, z, w; };
struct alignas(16) F4 { float x, y, z, w; };

union FragAB {
    bf16x16        v;
    unsigned short us[16];
    unsigned int   ui[8];
};
union FragC {
    f32x8 v;
    float f[8];
};

// Native conversion: clang lowers to v_cvt_pk_bf16_f32.
__device__ __forceinline__ void pack8(FragAB& fr, int base,
                                      const F4& a, const F4& b) {
    fr.v[base + 0] = (__bf16)a.x; fr.v[base + 1] = (__bf16)a.y;
    fr.v[base + 2] = (__bf16)a.z; fr.v[base + 3] = (__bf16)a.w;
    fr.v[base + 4] = (__bf16)b.x; fr.v[base + 5] = (__bf16)b.y;
    fr.v[base + 6] = (__bf16)b.z; fr.v[base + 7] = (__bf16)b.w;
}

__device__ __forceinline__ unsigned short f2bf_s(float f) {
    __bf16 h = (__bf16)f;
    return __builtin_bit_cast(unsigned short, h);
}

__device__ __forceinline__ void loadB(FragAB& b, const unsigned short* sW, int off) {
    *(U4*)(&b.ui[0]) = *(const U4*)(&sW[off]);
    *(U4*)(&b.ui[4]) = *(const U4*)(&sW[off + 16]);
}

// --------------------------------------------------------------------- zero
__global__ __launch_bounds__(256) void zero_kernel(F4* __restrict__ p, int n4) {
    int i = blockIdx.x * 256 + threadIdx.x;
    if (i < n4) { F4 z{0.f, 0.f, 0.f, 0.f}; p[i] = z; }
}

// --------------------------------------------------------------------- edge
// One workgroup = 16 waves = 256 edges. h = relu(ef@Wb+bb) fused with
// segment-max via atomicMax on non-negative float bits.
__global__ __launch_bounds__(512)
void edge_kernel(const float* __restrict__ feats,
                 const float* __restrict__ addInfo,
                 const int*   __restrict__ nidx,
                 const int*   __restrict__ segIds,
                 const float* __restrict__ Wb,
                 const float* __restrict__ bb,
                 unsigned int* __restrict__ pooled,
                 int E) {
    __shared__ unsigned short sW[H * KST_E];   // Wb^T, bf16, zero-padded K

    const int tid = threadIdx.x;
    // stage Wb^T: sW[n*KST_E + k] = bf16(Wb[k*H + n])
    for (int i = tid; i < (D + A) * H; i += 512) {
        int k = i >> 7, n = i & 127;
        sW[n * KST_E + k] = f2bf_s(Wb[i]);
    }
    for (int i = tid; i < H * (KST_E - (D + A)); i += 512) {
        int n = i / (KST_E - (D + A));
        int k = (D + A) + i % (KST_E - (D + A));
        sW[n * KST_E + k] = 0;
    }
    __syncthreads();

    const int wave = tid >> 5, lane = tid & 31;
    const int hi = lane >> 4, ln = lane & 15;
    const int strip = blockIdx.x * 256 + wave * 16;   // 16 edge rows per wave

    // A-gather source for this lane (row m = ln of the strip)
    const int eRow   = strip + ln;
    const int eClamp = eRow < E ? eRow : (E - 1);
    const int gidx   = nidx[eClamp];
    const float* __restrict__ rowp = feats + gidx * D;

    FragC acc[8];
#pragma unroll
    for (int nt = 0; nt < 8; ++nt)
#pragma unroll
        for (int r = 0; r < 8; ++r) acc[nt].f[r] = 0.f;

#pragma unroll
    for (int kt = 0; kt < 5; ++kt) {
        FragAB a;
        const int kb = kt * 32 + hi * 8;
        if (kt < 4) {
            F4 f0 = *(const F4*)(rowp + kb);
            F4 f1 = *(const F4*)(rowp + kb + 4);
            F4 f2 = *(const F4*)(rowp + kb + 16);
            F4 f3 = *(const F4*)(rowp + kb + 20);
            pack8(a, 0, f0, f1);
            pack8(a, 8, f2, f3);
        } else {
#pragma unroll
            for (int i = 0; i < 8; ++i) a.ui[i] = 0u;
            if (!hi) {   // K=128..131 come from add_info, rest zero
                F4 ai = *(const F4*)(addInfo + eClamp * A);
                a.v[0] = (__bf16)ai.x; a.v[1] = (__bf16)ai.y;
                a.v[2] = (__bf16)ai.z; a.v[3] = (__bf16)ai.w;
            }
        }
        // double-buffered B fragments: next LDS pair in flight under each WMMA
        FragAB bcur, bnxt;
        loadB(bcur, sW, ln * KST_E + kb);
#pragma unroll
        for (int nt = 0; nt < 8; ++nt) {
            if (nt < 7)
                loadB(bnxt, sW, ((nt + 1) * 16 + ln) * KST_E + kb);
            acc[nt].v = __builtin_amdgcn_wmma_f32_16x16x32_bf16(
                false, a.v, false, bcur.v, (short)0, acc[nt].v, false, false);
            bcur = bnxt;   // SSA rename after full unroll
        }
    }

    // epilogue: bias + relu + run-compressed segment max (ids sorted)
    int segArr[8];
#pragma unroll
    for (int r = 0; r < 8; ++r) {
        int e = strip + r + hi * 8;
        segArr[r] = (e < E) ? segIds[e] : -1;
    }
#pragma unroll
    for (int nt = 0; nt < 8; ++nt) {
        const int j = nt * 16 + ln;
        const float bbj = bb[j];
        int prevSeg = -1; float best = 0.f;
#pragma unroll
        for (int r = 0; r < 8; ++r) {
            if (segArr[r] >= 0) {
                float v = fmaxf(acc[nt].f[r] + bbj, 0.f);
                if (segArr[r] != prevSeg) {
                    if (prevSeg >= 0)
                        atomicMax(&pooled[prevSeg * H + j], __float_as_uint(best));
                    prevSeg = segArr[r];
                    best = v;
                } else {
                    best = fmaxf(best, v);
                }
            }
        }
        if (prevSeg >= 0)
            atomicMax(&pooled[prevSeg * H + j], __float_as_uint(best));
    }
}

// --------------------------------------------------------------------- proj
// dst = src + pooled @ Wo + bo   (pooled bits are non-negative floats)
__global__ __launch_bounds__(256)
void proj_kernel(const float* __restrict__ pooled,
                 const float* __restrict__ Wo,
                 const float* __restrict__ bo,
                 const float* __restrict__ src,
                 float* __restrict__ dst,
                 int N) {
    __shared__ unsigned short sW[D * KST_P];   // Wo^T

    const int tid = threadIdx.x;
    for (int i = tid; i < H * D; i += 256) {
        int h = i >> 7, j = i & 127;
        sW[j * KST_P + h] = f2bf_s(Wo[i]);
    }
    for (int i = tid; i < D * (KST_P - H); i += 256) {
        int j = i >> 3, k = H + (i & 7);
        sW[j * KST_P + k] = 0;
    }
    __syncthreads();

    const int wave = tid >> 5, lane = tid & 31;
    const int hi = lane >> 4, ln = lane & 15;
    const int mtiles = (N + 15) >> 4;
    const int mt = blockIdx.x * 8 + wave;
    if (mt >= mtiles) return;

    const int mBase = mt * 16;
    const int mRow  = (mBase + ln) < N ? (mBase + ln) : (N - 1);
    const float* __restrict__ rowp = pooled + mRow * H;

    FragC acc[8];
#pragma unroll
    for (int nt = 0; nt < 8; ++nt)
#pragma unroll
        for (int r = 0; r < 8; ++r) acc[nt].f[r] = 0.f;

#pragma unroll
    for (int kt = 0; kt < 4; ++kt) {
        const int kb = kt * 32 + hi * 8;
        FragAB a;
        F4 f0 = *(const F4*)(rowp + kb);
        F4 f1 = *(const F4*)(rowp + kb + 4);
        F4 f2 = *(const F4*)(rowp + kb + 16);
        F4 f3 = *(const F4*)(rowp + kb + 20);
        pack8(a, 0, f0, f1);
        pack8(a, 8, f2, f3);
        FragAB bcur, bnxt;
        loadB(bcur, sW, ln * KST_P + kb);
#pragma unroll
        for (int nt = 0; nt < 8; ++nt) {
            if (nt < 7)
                loadB(bnxt, sW, ((nt + 1) * 16 + ln) * KST_P + kb);
            acc[nt].v = __builtin_amdgcn_wmma_f32_16x16x32_bf16(
                false, a.v, false, bcur.v, (short)0, acc[nt].v, false, false);
            bcur = bnxt;
        }
    }

#pragma unroll
    for (int nt = 0; nt < 8; ++nt) {
        const int j = nt * 16 + ln;
        const float boj = bo[j];
#pragma unroll
        for (int r = 0; r < 8; ++r) {
            const int m = mBase + r + hi * 8;
            if (m < N)
                dst[m * D + j] = acc[nt].f[r] + boj + src[m * D + j];
        }
    }
}

// -------------------------------------------------------------------- final
// out[n] = dot(x[n,:], Wf) + bf   (one wave per row)
__global__ __launch_bounds__(256)
void final_kernel(const float* __restrict__ x,
                  const float* __restrict__ Wf,
                  const float* __restrict__ bf,
                  float* __restrict__ out,
                  int N) {
    const int tid = threadIdx.x;
    const int wave = tid >> 5, lane = tid & 31;
    const int n = blockIdx.x * 8 + wave;
    if (n >= N) return;
    F4 a = ((const F4*)(x + n * D))[lane];
    F4 w = ((const F4*)Wf)[lane];
    float s = a.x * w.x + a.y * w.y + a.z * w.z + a.w * w.w;
#pragma unroll
    for (int m = 16; m >= 1; m >>= 1)
        s += __shfl_xor(s, m, 32);
    if (lane == 0) out[n] = s + bf[0];
}

// ---------------------------------------------------------------------------
extern "C" void kernel_launch(void* const* d_in, const int* in_sizes, int n_in,
                              void* d_out, int out_size, void* d_ws, size_t ws_size,
                              hipStream_t stream) {
    const float* interp  = (const float*)d_in[0];
    const float* addInfo = (const float*)d_in[1];
    const int*   nidx    = (const int*)d_in[2];
    const int*   segIds  = (const int*)d_in[3];
    const float* Wb0 = (const float*)d_in[4];
    const float* bb0 = (const float*)d_in[5];
    const float* Wo0 = (const float*)d_in[6];
    const float* bo0 = (const float*)d_in[7];
    const float* Wb1 = (const float*)d_in[8];
    const float* bb1 = (const float*)d_in[9];
    const float* Wo1 = (const float*)d_in[10];
    const float* bo1 = (const float*)d_in[11];
    const float* Wf  = (const float*)d_in[12];
    const float* bfp = (const float*)d_in[13];

    const int N = in_sizes[0] / D;
    const int E = in_sizes[2];

    float* pooled = (float*)d_ws;              // N*H floats
    float* xws    = pooled + (size_t)N * H;    // N*D floats
    unsigned int* pooledU = (unsigned int*)pooled;

    const int n4      = (N * H) / 4;
    const int zeroGrd = (n4 + 255) / 256;
    const int edgeGrd = (E + 255) / 256;
    const int projGrd = (((N + 15) / 16) + 7) / 8;
    const int finGrd  = (N + 7) / 8;

    // ---- block 0 ----
    zero_kernel<<<zeroGrd, 256, 0, stream>>>((F4*)pooled, n4);
    edge_kernel<<<edgeGrd, 512, 0, stream>>>(interp, addInfo, nidx, segIds,
                                             Wb0, bb0, pooledU, E);
    proj_kernel<<<projGrd, 256, 0, stream>>>(pooled, Wo0, bo0, interp, xws, N);

    // ---- block 1 ----
    zero_kernel<<<zeroGrd, 256, 0, stream>>>((F4*)pooled, n4);
    edge_kernel<<<edgeGrd, 512, 0, stream>>>(xws, addInfo, nidx, segIds,
                                             Wb1, bb1, pooledU, E);
    proj_kernel<<<projGrd, 256, 0, stream>>>(pooled, Wo1, bo1, xws, xws, N);

    // ---- final score ----
    final_kernel<<<finGrd, 256, 0, stream>>>(xws, Wf, bfp, (float*)d_out, N);
}
